// TemporalEncoderMamba_89129161326946
// MI455X (gfx1250) — matrix operand order repaired and checked by reading, not compiled, not run on hardware
//
#include <hip/hip_runtime.h>
#include <hip/hip_bf16.h>
#include <math.h>

typedef __attribute__((ext_vector_type(16))) __bf16 v16bf_t;
typedef __attribute__((ext_vector_type(8)))  __bf16 v8bf_t;
typedef __attribute__((ext_vector_type(8)))  float  v8f_t;

#define B_DIM 16
#define S_DIM 4096
#define D_DIM 256
#define H_DIM 512

__device__ __forceinline__ float sigmoidf_(float x) { return 1.0f / (1.0f + __expf(-x)); }

__device__ __forceinline__ v8f_t v8f_zero() {
  v8f_t v;
#pragma unroll
  for (int i = 0; i < 8; ++i) v[i] = 0.0f;
  return v;
}

// ---------------------------------------------------------------------------
// f32 -> bf16 weight pre-conversion (tiny, one-time per launch).
// ---------------------------------------------------------------------------
__global__ __launch_bounds__(256)
void cvt_kernel(const float* __restrict__ src, __bf16* __restrict__ dst, int n)
{
  const int i = (blockIdx.x * 256 + threadIdx.x) * 4;
  if (i + 3 < n) {
#pragma unroll
    for (int j = 0; j < 4; ++j) dst[i + j] = (__bf16)src[i + j];
  }
}

// ---------------------------------------------------------------------------
// Projection GEMM: out[n, o] = act( X[n,:] . W[o,:] + bias[o] )
//   mode 0: tanh(.) * mask[n]   -> h buffer
//   mode 1: sigmoid(.)          -> g buffer
// Wave computes 16x64 (4 WMMA accumulators). All 4 B tiles are loaded as bf16
// from ONE base pointer with compile-time byte offsets (folded into the load's
// 24-bit IOFFSET), and all loads are issued before the 4 back-to-back WMMAs.
// ---------------------------------------------------------------------------
__global__ __launch_bounds__(128)
void proj_kernel(const float* __restrict__ X, const __bf16* __restrict__ Wb,
                 const float* __restrict__ bias, const int* __restrict__ mask,
                 float* __restrict__ out, int mode)
{
  const int lane = threadIdx.x & 31;
  const int wave = threadIdx.x >> 5;
  const int m    = lane & 15;
  const int half = lane >> 4;
  const int row0 = blockIdx.x * 16;
  const int col0 = (blockIdx.y * 4 + wave) * 64;    // 4 x 16 columns per wave

  const float*  arow  = X + (size_t)(row0 + m) * D_DIM;
  const __bf16* bbase = Wb + (size_t)(col0 + m) * D_DIM + half * 16;

  v8f_t acc[4];
#pragma unroll
  for (int t = 0; t < 4; ++t) acc[t] = v8f_zero();

#pragma unroll
  for (int kb = 0; kb < D_DIM; kb += 32) {
    // A 16x32 bf16: lane row m; elems 0..7 -> K = kb + half*16 + i,
    //               elems 8..15 -> K = kb + half*16 + 8 + i
    v8f_t a_lo = *(const v8f_t*)(arow + kb + half * 16);
    v8f_t a_hi = *(const v8f_t*)(arow + kb + half * 16 + 8);
    v16bf_t a;
#pragma unroll
    for (int i = 0; i < 8; ++i) {
      a[i]     = (__bf16)a_lo[i];
      a[i + 8] = (__bf16)a_hi[i];
    }
    // B 32x16 bf16: lane col; elem e -> K = kb + half*16 + e; tile t at a
    // constant offset t*16*D from the base (fits signed 24-bit IOFFSET).
    v16bf_t bt[4];
#pragma unroll
    for (int t = 0; t < 4; ++t) {
      v8bf_t blo = *(const v8bf_t*)(bbase + kb + t * (16 * D_DIM));
      v8bf_t bhi = *(const v8bf_t*)(bbase + kb + t * (16 * D_DIM) + 8);
#pragma unroll
      for (int i = 0; i < 8; ++i) { bt[t][i] = blo[i]; bt[t][i + 8] = bhi[i]; }
    }
#pragma unroll
    for (int t = 0; t < 4; ++t) {
      acc[t] = __builtin_amdgcn_wmma_f32_16x16x32_bf16(false, a, false, bt[t],
                                                       (short)0, acc[t], false, false);
    }
  }

  // C/D layout: VGPR r -> row (r + 8*half), col (lane&15)
  float mf[8];
#pragma unroll
  for (int r = 0; r < 8; ++r)
    mf[r] = (mode == 0) ? (float)mask[row0 + r + 8 * half] : 1.0f;

#pragma unroll
  for (int t = 0; t < 4; ++t) {
    const int col = col0 + t * 16 + m;
    const float bcol = bias[col];
    float* obase = out + (size_t)(row0 + 8 * half) * H_DIM + col;
#pragma unroll
    for (int r = 0; r < 8; ++r) {
      float v = acc[t][r] + bcol;
      v = (mode == 0) ? tanhf(v) * mf[r] : sigmoidf_(v);
      obase[r * H_DIM] = v;
    }
  }
}

// ---------------------------------------------------------------------------
// Pointwise GEMM with fused depthwise conv (K=3, same-pad, per-batch) on the
// A operand, plus fused residual + SiLU epilogue. Wave computes 16x64 so the
// conv A-operand computation is amortized over 4 WMMAs; B tiles at constant
// immediate offsets, loads grouped ahead of the WMMAs.
//   hc[n,k] = dw_b[k] + sum_t dw_w[k,t] * h[n + t - 1, k]   (zero pad at batch edges)
//   u[n,o]  = silu( sum_k hc[n,k]*pw_w[o,k] + pw_b[o] + h[n,o] )
// ---------------------------------------------------------------------------
__global__ __launch_bounds__(128)
void pw_kernel(const float* __restrict__ h, const float* __restrict__ dw_w,
               const float* __restrict__ dw_b, const __bf16* __restrict__ pw_wb,
               const float* __restrict__ pw_b, float* __restrict__ u)
{
  __shared__ float s_dww[H_DIM * 3];
  __shared__ float s_dwb[H_DIM];
  for (int i = threadIdx.x; i < H_DIM * 3; i += blockDim.x) s_dww[i] = dw_w[i];
  for (int i = threadIdx.x; i < H_DIM; i += blockDim.x)     s_dwb[i] = dw_b[i];
  __syncthreads();

  const int lane = threadIdx.x & 31;
  const int wave = threadIdx.x >> 5;
  const int m    = lane & 15;
  const int half = lane >> 4;
  const int row0 = blockIdx.x * 16;                 // 16 | S, tiles never straddle batches
  const int col0 = (blockIdx.y * 4 + wave) * 64;

  const int n = row0 + m;
  const int s = n & (S_DIM - 1);
  const bool has_m1 = (s > 0);
  const bool has_p1 = (s < S_DIM - 1);

  const float*  hrow  = h + (size_t)n * H_DIM;
  const __bf16* bbase = pw_wb + (size_t)(col0 + m) * H_DIM + half * 16;

  v8f_t acc[4];
#pragma unroll
  for (int t = 0; t < 4; ++t) acc[t] = v8f_zero();

#pragma unroll 2
  for (int kb = 0; kb < H_DIM; kb += 32) {
    v16bf_t a;
#pragma unroll
    for (int gsel = 0; gsel < 2; ++gsel) {
      const int k0 = kb + half * 16 + gsel * 8;
      v8f_t h0 = *(const v8f_t*)(hrow + k0);
      v8f_t hm = v8f_zero();
      v8f_t hp = v8f_zero();
      if (has_m1) hm = *(const v8f_t*)(hrow - H_DIM + k0);
      if (has_p1) hp = *(const v8f_t*)(hrow + H_DIM + k0);
#pragma unroll
      for (int i = 0; i < 8; ++i) {
        const int k = k0 + i;
        float c = s_dwb[k] + s_dww[k * 3 + 0] * hm[i]
                           + s_dww[k * 3 + 1] * h0[i]
                           + s_dww[k * 3 + 2] * hp[i];
        a[gsel * 8 + i] = (__bf16)c;
      }
    }
    v16bf_t bt[4];
#pragma unroll
    for (int t = 0; t < 4; ++t) {
      v8bf_t blo = *(const v8bf_t*)(bbase + kb + t * (16 * H_DIM));
      v8bf_t bhi = *(const v8bf_t*)(bbase + kb + t * (16 * H_DIM) + 8);
#pragma unroll
      for (int i = 0; i < 8; ++i) { bt[t][i] = blo[i]; bt[t][i + 8] = bhi[i]; }
    }
#pragma unroll
    for (int t = 0; t < 4; ++t) {
      acc[t] = __builtin_amdgcn_wmma_f32_16x16x32_bf16(false, a, false, bt[t],
                                                       (short)0, acc[t], false, false);
    }
  }

#pragma unroll
  for (int t = 0; t < 4; ++t) {
    const int col = col0 + t * 16 + m;
    const float bcol = pw_b[col];
    const float* rbase = h + (size_t)(row0 + 8 * half) * H_DIM + col;
    float*       obase = u + (size_t)(row0 + 8 * half) * H_DIM + col;
#pragma unroll
    for (int r = 0; r < 8; ++r) {
      float v = acc[t][r] + bcol + rbase[r * H_DIM];
      v = v * sigmoidf_(v);                          // SiLU
      obase[r * H_DIM] = v;
    }
  }
}

// ---------------------------------------------------------------------------
// Sequential gated scan: one thread per (b, channel); coalesced stride-H loads.
// ---------------------------------------------------------------------------
__global__ __launch_bounds__(512)
void scan_kernel(const float* __restrict__ u, const float* __restrict__ g,
                 const int* __restrict__ mask, const float* __restrict__ A_log,
                 const float* __restrict__ Bp, const float* __restrict__ Cp,
                 float* __restrict__ y)
{
  const int b  = blockIdx.x;
  const int ch = threadIdx.x;
  const float al  = A_log[ch];
  const float sp  = (al > 20.0f) ? al : log1pf(__expf(al));   // softplus
  const float a_e = __expf(-sp);
  const float bp  = Bp[ch];
  const float cp  = Cp[ch];

  const float* up = u + (size_t)b * S_DIM * H_DIM + ch;
  const float* gp = g + (size_t)b * S_DIM * H_DIM + ch;
  const int*   mp = mask + (size_t)b * S_DIM;
  float*       yp = y + (size_t)b * S_DIM * H_DIM + ch;

  float state = 0.0f, last = 0.0f;
  for (int s = 0; s < S_DIM; ++s) {
    const float ut = up[(size_t)s * H_DIM];
    const float gt = gp[(size_t)s * H_DIM];
    const float ns = gt * (a_e * state) + (1.0f - gt) * (bp * ut);
    if (mp[s] != 0) {
      state = ns;
      last  = state * cp;
    }
    yp[(size_t)s * H_DIM] = last;
  }
}

// ---------------------------------------------------------------------------
// LayerNorm over H: one wave32 per row, shuffle reduction.
// ---------------------------------------------------------------------------
__global__ __launch_bounds__(256)
void ln_kernel(const float* __restrict__ y, const float* __restrict__ ln_g,
               const float* __restrict__ ln_b, float* __restrict__ out)
{
  const int lane = threadIdx.x & 31;
  const int wave = threadIdx.x >> 5;
  const size_t row = (size_t)blockIdx.x * 8 + wave;
  const float* p = y + row * H_DIM;

  float vals[16];
  float sum = 0.0f, sq = 0.0f;
#pragma unroll
  for (int i = 0; i < 16; ++i) {
    const float v = p[lane + i * 32];
    vals[i] = v;
    sum += v;
    sq  += v * v;
  }
#pragma unroll
  for (int off = 16; off > 0; off >>= 1) {
    sum += __shfl_xor(sum, off, 32);
    sq  += __shfl_xor(sq,  off, 32);
  }
  const float mu  = sum * (1.0f / H_DIM);
  const float var = sq * (1.0f / H_DIM) - mu * mu;
  const float inv = rsqrtf(var + 1e-5f);

  float* o = out + row * H_DIM;
#pragma unroll
  for (int i = 0; i < 16; ++i) {
    const int c = lane + i * 32;
    o[c] = (vals[i] - mu) * inv * ln_g[c] + ln_b[c];
  }
}

// ---------------------------------------------------------------------------
extern "C" void kernel_launch(void* const* d_in, const int* in_sizes, int n_in,
                              void* d_out, int out_size, void* d_ws, size_t ws_size,
                              hipStream_t stream)
{
  (void)in_sizes; (void)n_in; (void)out_size; (void)ws_size;

  const float* inputs = (const float*)d_in[0];
  const int*   mask   = (const int*)d_in[1];
  const float* in_w   = (const float*)d_in[2];
  const float* in_b   = (const float*)d_in[3];
  const float* gate_w = (const float*)d_in[4];
  const float* gate_b = (const float*)d_in[5];
  const float* dw_w   = (const float*)d_in[6];
  const float* dw_b   = (const float*)d_in[7];
  const float* pw_w   = (const float*)d_in[8];
  const float* pw_b   = (const float*)d_in[9];
  const float* ln_g   = (const float*)d_in[10];
  const float* ln_b   = (const float*)d_in[11];
  const float* A_log  = (const float*)d_in[12];
  const float* Bp     = (const float*)d_in[13];
  const float* Cp     = (const float*)d_in[14];

  const size_t NT = (size_t)B_DIM * S_DIM * H_DIM;   // 33,554,432 elements
  float* h = (float*)d_ws;        // [B,S,H], later reused as scan output y
  float* g = h + NT;              // [B,S,H]
  float* u = g + NT;              // [B,S,H]
  __bf16* in_wb   = (__bf16*)(u + NT);               // [H,D] bf16
  __bf16* gate_wb = in_wb + (size_t)H_DIM * D_DIM;   // [H,D] bf16
  __bf16* pw_wb   = gate_wb + (size_t)H_DIM * D_DIM; // [H,H] bf16

  // One-time weight conversion (deterministic; re-run every call)
  const int nW  = H_DIM * D_DIM;                     // 131072
  const int nPW = H_DIM * H_DIM;                     // 262144
  cvt_kernel<<<nW  / 1024, 256, 0, stream>>>(in_w,   in_wb,   nW);
  cvt_kernel<<<nW  / 1024, 256, 0, stream>>>(gate_w, gate_wb, nW);
  cvt_kernel<<<nPW / 1024, 256, 0, stream>>>(pw_w,   pw_wb,   nPW);

  dim3 blk(128);
  dim3 grd((B_DIM * S_DIM) / 16, H_DIM / 256);       // 4 waves x 64 cols = 256 cols/block

  proj_kernel<<<grd, blk, 0, stream>>>(inputs, in_wb,   in_b,   mask, h, 0);
  proj_kernel<<<grd, blk, 0, stream>>>(inputs, gate_wb, gate_b, mask, g, 1);
  pw_kernel<<<grd, blk, 0, stream>>>(h, dw_w, dw_b, pw_wb, pw_b, u);
  scan_kernel<<<B_DIM, H_DIM, 0, stream>>>(u, g, mask, A_log, Bp, Cp, h);
  ln_kernel<<<(B_DIM * S_DIM) / 8, 256, 0, stream>>>(h, ln_g, ln_b, (float*)d_out);
}
